// GINCombined_13262859010607
// MI455X (gfx1250) — compile-verified, matrix operand order checked by reading, not run on hardware
//
#include <hip/hip_runtime.h>
#include <hip/hip_bf16.h>

// ---------------------------------------------------------------------------
// GIN graph network for MI455X (gfx1250, wave32, WMMA).
// GEMMs use v_wmma_f32_16x16x32_f16 with f32 accumulation; weights are
// pre-packed into B-fragment order (one contiguous 32B load per lane per
// fragment). Edge aggregation / pooling are bandwidth-bound scatter kernels.
// ---------------------------------------------------------------------------

#define N_NODES 100000
#define N_EDGES 1600000
#define G_GRAPHS 128
#define D_FEATC 240
#define EMB_DIM 16
#define HDIM 256
#define PHDIM 128
#define GFDIM 32
#define LN_EPS 1e-5f

typedef __attribute__((ext_vector_type(16))) _Float16 v16h;
typedef __attribute__((ext_vector_type(8)))  _Float16 v8h;
typedef __attribute__((ext_vector_type(8)))  float    v8f;

union AFrag { v16h v; v8h h[2]; };

// WMMA fragment index helpers (16-bit A 16x32 layout, ISA 7.12.2):
// lane L (m = L&15, kbase = (L&16)?8:0), half j -> k = (j<8?0:16)+kbase+(j&7)
// => halves j=0..7 and j=8..15 are each 8 contiguous k values.

// ---------------------------------------------------------------------------
// Pack a row-major f32 weight [K][Ncols] into f16 B-fragment order:
// wp[(((ct*K/32 + ks))*32 + lane)*16 + j] = w[k][n]
//   n = ct*16 + (lane&15)
//   k = ks*32 + (j<8?0:16) + ((lane&16)?8:0) + (j&7)
// ---------------------------------------------------------------------------
__global__ void pack_weight_kernel(const float* __restrict__ w,
                                   _Float16* __restrict__ wp,
                                   int K, int Ncols) {
  int total = K * Ncols;
  int kc = K >> 5;
  for (int idx = blockIdx.x * blockDim.x + threadIdx.x; idx < total;
       idx += gridDim.x * blockDim.x) {
    int j    = idx & 15;
    int lane = (idx >> 4) & 31;
    int rest = idx >> 9;
    int ks   = rest % kc;
    int ct   = rest / kc;
    int n = ct * 16 + (lane & 15);
    int k = ks * 32 + ((j >> 3) ? 16 : 0) + ((lane & 16) ? 8 : 0) + (j & 7);
    wp[idx] = (_Float16)w[k * Ncols + n];
  }
}

// h[i] = concat(feats[i], emb_table[node_type[i]])
__global__ void embed_concat_kernel(const float* __restrict__ x,
                                    const float* __restrict__ emb,
                                    float* __restrict__ h) {
  int idx = blockIdx.x * blockDim.x + threadIdx.x;
  if (idx >= N_NODES * HDIM) return;
  int i = idx >> 8;
  int c = idx & 255;
  float v;
  if (c < D_FEATC) {
    v = x[i * (D_FEATC + 1) + 1 + c];
  } else {
    int nt = (int)x[i * (D_FEATC + 1)];
    v = emb[nt * EMB_DIM + (c - D_FEATC)];
  }
  h[idx] = v;
}

__global__ void copy4_kernel(const float4* __restrict__ src,
                             float4* __restrict__ dst, int n4) {
  int idx = blockIdx.x * blockDim.x + threadIdx.x;
  if (idx < n4) dst[idx] = src[idx];
}

// z[dst] += h[src]  (64 lanes per edge, float4 per lane, 4 edges per block)
__global__ void edge_scatter_kernel(const int* __restrict__ srcIdx,
                                    const int* __restrict__ dstIdx,
                                    const float* __restrict__ h,
                                    float* __restrict__ z) {
  int t = threadIdx.x;
  int e = blockIdx.x * 4 + (t >> 6);
  if (e >= N_EDGES) return;
  int c0 = (t & 63) * 4;
  int s = srcIdx[e];
  int d = dstIdx[e];
  float4 hv = *reinterpret_cast<const float4*>(h + s * HDIM + c0);
  float* zp = z + d * HDIM + c0;
  atomicAdd(zp + 0, hv.x);
  atomicAdd(zp + 1, hv.y);
  atomicAdd(zp + 2, hv.z);
  atomicAdd(zp + 3, hv.w);
}

// ---------------------------------------------------------------------------
// GEMM1: zout_f16 = relu(zin_f32 @ Wpack + bias).  [N,256]x[256,256]
// Block = 32 rows x 256 cols. 8 waves; wave -> (rowtile = w>>2, colgroup = w&3)
// A staged f32->f16 into LDS (row stride 264 halves to de-bank-conflict).
// ---------------------------------------------------------------------------
__global__ void __launch_bounds__(256)
gemm_relu_f16_kernel(const float* __restrict__ zin,
                     const _Float16* __restrict__ wp,
                     const float* __restrict__ bias,
                     _Float16* __restrict__ zout) {
  __shared__ _Float16 a_tile[32 * 264];
  const int t = threadIdx.x;
  const int row0 = blockIdx.x * 32;

  for (int idx = t; idx < 32 * HDIM; idx += 256) {
    int r = idx >> 8, c = idx & 255;
    a_tile[r * 264 + c] = (_Float16)zin[(row0 + r) * HDIM + c];
  }
  __syncthreads();

  const int wave = t >> 5, lane = t & 31;
  const int rt = wave >> 2;
  const int cg = wave & 3;
  const int mloc = rt * 16 + (lane & 15);
  const int kb = (lane & 16) ? 8 : 0;

  v8f acc0 = {}, acc1 = {}, acc2 = {}, acc3 = {};
#pragma unroll
  for (int ks = 0; ks < 8; ++ks) {
    AFrag af;
    const _Float16* ap = &a_tile[mloc * 264 + ks * 32 + kb];
    af.h[0] = *reinterpret_cast<const v8h*>(ap);
    af.h[1] = *reinterpret_cast<const v8h*>(ap + 16);
    const _Float16* bp = wp + ((cg * 4 * 8 + ks) * 32 + lane) * 16;
    v16h b0 = *reinterpret_cast<const v16h*>(bp);
    v16h b1 = *reinterpret_cast<const v16h*>(bp + 1 * 8 * 32 * 16);
    v16h b2 = *reinterpret_cast<const v16h*>(bp + 2 * 8 * 32 * 16);
    v16h b3 = *reinterpret_cast<const v16h*>(bp + 3 * 8 * 32 * 16);
    acc0 = __builtin_amdgcn_wmma_f32_16x16x32_f16(false, af.v, false, b0, (short)0, acc0, false, false);
    acc1 = __builtin_amdgcn_wmma_f32_16x16x32_f16(false, af.v, false, b1, (short)0, acc1, false, false);
    acc2 = __builtin_amdgcn_wmma_f32_16x16x32_f16(false, af.v, false, b2, (short)0, acc2, false, false);
    acc3 = __builtin_amdgcn_wmma_f32_16x16x32_f16(false, af.v, false, b3, (short)0, acc3, false, false);
  }

  const int mout = rt * 16 + ((lane & 16) ? 8 : 0);
  v8f accs[4] = {acc0, acc1, acc2, acc3};
#pragma unroll
  for (int ct = 0; ct < 4; ++ct) {
    int n = (cg * 4 + ct) * 16 + (lane & 15);
    float bv = bias[n];
#pragma unroll
    for (int r = 0; r < 8; ++r) {
      float v = accs[ct][r] + bv;
      v = v > 0.f ? v : 0.f;
      zout[(row0 + mout + r) * HDIM + n] = (_Float16)v;
    }
  }
}

// ---------------------------------------------------------------------------
// GEMM2 + LayerNorm + ReLU + residual:  h += relu(LN(a_f16 @ Wpack + bias))
// A fragments read directly from global f16 (output of GEMM1).
// Block owns the full 32x256 output tile -> LN reduction is block-local.
// ---------------------------------------------------------------------------
__global__ void __launch_bounds__(256)
gemm_ln_res_kernel(const _Float16* __restrict__ a,
                   const _Float16* __restrict__ wp,
                   const float* __restrict__ bias,
                   const float* __restrict__ ln_g,
                   const float* __restrict__ ln_b,
                   float* __restrict__ h) {
  __shared__ float zt[32 * 264];
  __shared__ float red[32 * 8];
  const int t = threadIdx.x;
  const int row0 = blockIdx.x * 32;
  const int wave = t >> 5, lane = t & 31;
  const int rt = wave >> 2, cg = wave & 3;
  const int mloc = rt * 16 + (lane & 15);
  const int kb = (lane & 16) ? 8 : 0;

  const _Float16* arow = a + (row0 + mloc) * HDIM;

  v8f acc0 = {}, acc1 = {}, acc2 = {}, acc3 = {};
#pragma unroll
  for (int ks = 0; ks < 8; ++ks) {
    AFrag af;
    const _Float16* ap = arow + ks * 32 + kb;
    af.h[0] = *reinterpret_cast<const v8h*>(ap);
    af.h[1] = *reinterpret_cast<const v8h*>(ap + 16);
    if (ks < 7) __builtin_prefetch(arow + (ks + 1) * 32 + kb, 0, 0);
    const _Float16* bp = wp + ((cg * 4 * 8 + ks) * 32 + lane) * 16;
    v16h b0 = *reinterpret_cast<const v16h*>(bp);
    v16h b1 = *reinterpret_cast<const v16h*>(bp + 1 * 8 * 32 * 16);
    v16h b2 = *reinterpret_cast<const v16h*>(bp + 2 * 8 * 32 * 16);
    v16h b3 = *reinterpret_cast<const v16h*>(bp + 3 * 8 * 32 * 16);
    acc0 = __builtin_amdgcn_wmma_f32_16x16x32_f16(false, af.v, false, b0, (short)0, acc0, false, false);
    acc1 = __builtin_amdgcn_wmma_f32_16x16x32_f16(false, af.v, false, b1, (short)0, acc1, false, false);
    acc2 = __builtin_amdgcn_wmma_f32_16x16x32_f16(false, af.v, false, b2, (short)0, acc2, false, false);
    acc3 = __builtin_amdgcn_wmma_f32_16x16x32_f16(false, af.v, false, b3, (short)0, acc3, false, false);
  }

  const int mout = rt * 16 + ((lane & 16) ? 8 : 0);
  v8f accs[4] = {acc0, acc1, acc2, acc3};
#pragma unroll
  for (int ct = 0; ct < 4; ++ct) {
    int n = (cg * 4 + ct) * 16 + (lane & 15);
    float bv = bias[n];
#pragma unroll
    for (int r = 0; r < 8; ++r) {
      zt[(mout + r) * 264 + n] = accs[ct][r] + bv;
    }
  }
  __syncthreads();

  // LayerNorm: 8 threads per row, 32 cols each.
  const int row = t >> 3, seg = t & 7;
  float ssum = 0.f;
  for (int c = 0; c < 32; ++c) ssum += zt[row * 264 + seg * 32 + c];
  red[row * 8 + seg] = ssum;
  __syncthreads();
  float mean = 0.f;
  for (int k = 0; k < 8; ++k) mean += red[row * 8 + k];
  mean *= (1.f / HDIM);
  __syncthreads();
  float vsum = 0.f;
  for (int c = 0; c < 32; ++c) {
    float d = zt[row * 264 + seg * 32 + c] - mean;
    vsum += d * d;
  }
  red[row * 8 + seg] = vsum;
  __syncthreads();
  float var = 0.f;
  for (int k = 0; k < 8; ++k) var += red[row * 8 + k];
  var *= (1.f / HDIM);
  float inv = rsqrtf(var + LN_EPS);
  for (int c = 0; c < 32; ++c) {
    int col = seg * 32 + c;
    float zn = (zt[row * 264 + col] - mean) * inv * ln_g[col] + ln_b[col];
    zn = zn > 0.f ? zn : 0.f;
    int gi = (row0 + row) * HDIM + col;
    h[gi] = h[gi] + zn;
  }
}

// ---------------------------------------------------------------------------
// Gate: gate[n] = relu(h @ Wg1 + bg1) . wg2 + bg2   (256->128->1, fused)
// ---------------------------------------------------------------------------
__global__ void __launch_bounds__(256)
gate_kernel(const float* __restrict__ h,
            const _Float16* __restrict__ wp,
            const float* __restrict__ b1,
            const float* __restrict__ w2,
            const float* __restrict__ b2,
            float* __restrict__ gate) {
  __shared__ _Float16 a_tile[32 * 264];
  __shared__ float g_tile[32 * 136];
  __shared__ float red[32 * 8];
  const int t = threadIdx.x;
  const int row0 = blockIdx.x * 32;

  for (int idx = t; idx < 32 * HDIM; idx += 256) {
    int r = idx >> 8, c = idx & 255;
    a_tile[r * 264 + c] = (_Float16)h[(row0 + r) * HDIM + c];
  }
  __syncthreads();

  const int wave = t >> 5, lane = t & 31;
  const int rt = wave >> 2, cg = wave & 3;  // 8 col tiles: cg*2, cg*2+1
  const int mloc = rt * 16 + (lane & 15);
  const int kb = (lane & 16) ? 8 : 0;

  v8f acc0 = {}, acc1 = {};
#pragma unroll
  for (int ks = 0; ks < 8; ++ks) {
    AFrag af;
    const _Float16* ap = &a_tile[mloc * 264 + ks * 32 + kb];
    af.h[0] = *reinterpret_cast<const v8h*>(ap);
    af.h[1] = *reinterpret_cast<const v8h*>(ap + 16);
    const _Float16* bp = wp + ((cg * 2 * 8 + ks) * 32 + lane) * 16;
    v16h b0 = *reinterpret_cast<const v16h*>(bp);
    v16h b1v = *reinterpret_cast<const v16h*>(bp + 8 * 32 * 16);
    acc0 = __builtin_amdgcn_wmma_f32_16x16x32_f16(false, af.v, false, b0, (short)0, acc0, false, false);
    acc1 = __builtin_amdgcn_wmma_f32_16x16x32_f16(false, af.v, false, b1v, (short)0, acc1, false, false);
  }

  const int mout = rt * 16 + ((lane & 16) ? 8 : 0);
  v8f accs[2] = {acc0, acc1};
#pragma unroll
  for (int ct = 0; ct < 2; ++ct) {
    int n = (cg * 2 + ct) * 16 + (lane & 15);
    float bv = b1[n];
#pragma unroll
    for (int r = 0; r < 8; ++r) {
      float v = accs[ct][r] + bv;
      g_tile[(mout + r) * 136 + n] = v > 0.f ? v : 0.f;
    }
  }
  __syncthreads();

  // dot with w2[128] per row
  const int row = t >> 3, seg = t & 7;
  float s = 0.f;
  for (int c = 0; c < 16; ++c) {
    int col = seg * 16 + c;
    s += g_tile[row * 136 + col] * w2[col];
  }
  red[row * 8 + seg] = s;
  __syncthreads();
  if (seg == 0) {
    float tot = b2[0];
    for (int k = 0; k < 8; ++k) tot += red[row * 8 + k];
    gate[row0 + row] = tot;
  }
}

__global__ void pool_init_kernel(float* __restrict__ pooled,
                                 float* __restrict__ m,
                                 float* __restrict__ s) {
  int idx = blockIdx.x * blockDim.x + threadIdx.x;
  if (idx < G_GRAPHS * HDIM) pooled[idx] = 0.f;
  if (idx < G_GRAPHS) { m[idx] = -3.0e38f; s[idx] = 0.f; }
}

__device__ inline void atomicMaxF(float* addr, float val) {
  int* ai = reinterpret_cast<int*>(addr);
  int old = __float_as_int(*addr);
  while (__int_as_float(old) < val) {
    int assumed = old;
    old = atomicCAS(ai, assumed, __float_as_int(val));
    if (old == assumed) break;
  }
}

__global__ void seg_max_kernel(const float* __restrict__ gate,
                               const int* __restrict__ batch,
                               float* __restrict__ m) {
  int n = blockIdx.x * blockDim.x + threadIdx.x;
  if (n < N_NODES) atomicMaxF(&m[batch[n]], gate[n]);
}

__global__ void seg_exp_kernel(const float* __restrict__ gate,
                               const int* __restrict__ batch,
                               const float* __restrict__ m,
                               float* __restrict__ a,
                               float* __restrict__ s) {
  int n = blockIdx.x * blockDim.x + threadIdx.x;
  if (n < N_NODES) {
    int g = batch[n];
    float av = expf(gate[n] - m[g]);
    a[n] = av;
    atomicAdd(&s[g], av);
  }
}

__global__ void pool_scatter_kernel(const float* __restrict__ h,
                                    const float* __restrict__ a,
                                    const float* __restrict__ s,
                                    const int* __restrict__ batch,
                                    float* __restrict__ pooled) {
  int idx = blockIdx.x * blockDim.x + threadIdx.x;
  if (idx >= N_NODES * (HDIM / 4)) return;
  int n = idx >> 6;
  int c0 = (idx & 63) * 4;
  int g = batch[n];
  float w = a[n] / s[g];
  float4 hv = *reinterpret_cast<const float4*>(h + n * HDIM + c0);
  float* pp = pooled + g * HDIM + c0;
  atomicAdd(pp + 0, w * hv.x);
  atomicAdd(pp + 1, w * hv.y);
  atomicAdd(pp + 2, w * hv.z);
  atomicAdd(pp + 3, w * hv.w);
}

// out = relu(concat(pooled, gf) @ W1 + b1) @ W2 + b2   (G=128 rows; tiny)
__global__ void __launch_bounds__(128)
cls_kernel(const float* __restrict__ pooled,
           const float* __restrict__ gf,
           const float* __restrict__ w1,
           const float* __restrict__ b1,
           const float* __restrict__ w2,
           const float* __restrict__ b2,
           float* __restrict__ out) {
  __shared__ float hid[PHDIM];
  int g = blockIdx.x, j = threadIdx.x;
  float acc = b1[j];
  for (int c = 0; c < HDIM; ++c) acc += pooled[g * HDIM + c] * w1[c * PHDIM + j];
  for (int c = 0; c < GFDIM; ++c) acc += gf[g * GFDIM + c] * w1[(HDIM + c) * PHDIM + j];
  hid[j] = acc > 0.f ? acc : 0.f;
  __syncthreads();
  if (j < 2) {
    float o = b2[j];
    for (int k = 0; k < PHDIM; ++k) o += hid[k] * w2[k * 2 + j];
    out[g * 2 + j] = o;
  }
}

// ---------------------------------------------------------------------------
extern "C" void kernel_launch(void* const* d_in, const int* in_sizes, int n_in,
                              void* d_out, int out_size, void* d_ws, size_t ws_size,
                              hipStream_t stream) {
  const float* x     = (const float*)d_in[0];
  const int*   edges = (const int*)d_in[1];
  const int*   batch = (const int*)d_in[2];
  const float* gfeat = (const float*)d_in[3];
  const float* emb   = (const float*)d_in[4];
  const float* l0_w1 = (const float*)d_in[5];
  const float* l0_b1 = (const float*)d_in[6];
  const float* l0_w2 = (const float*)d_in[7];
  const float* l0_b2 = (const float*)d_in[8];
  const float* ln0_g = (const float*)d_in[9];
  const float* ln0_b = (const float*)d_in[10];
  const float* l1_w1 = (const float*)d_in[11];
  const float* l1_b1 = (const float*)d_in[12];
  const float* l1_w2 = (const float*)d_in[13];
  const float* l1_b2 = (const float*)d_in[14];
  const float* ln1_g = (const float*)d_in[15];
  const float* ln1_b = (const float*)d_in[16];
  const float* gw1   = (const float*)d_in[17];
  const float* gb1   = (const float*)d_in[18];
  const float* gw2   = (const float*)d_in[19];
  const float* gb2   = (const float*)d_in[20];
  const float* cw1   = (const float*)d_in[21];
  const float* cb1   = (const float*)d_in[22];
  const float* cw2   = (const float*)d_in[23];
  const float* cb2   = (const float*)d_in[24];
  float* out = (float*)d_out;

  const int* src = edges;            // edge_index[0]
  const int* dst = edges + N_EDGES;  // edge_index[1]

  // workspace carve-up
  char* ws = (char*)d_ws;
  size_t off = 0;
  auto alloc = [&](size_t bytes) -> void* {
    off = (off + 255) & ~(size_t)255;
    void* p = ws + off;
    off += bytes;
    return p;
  };
  float*    h      = (float*)alloc((size_t)N_NODES * HDIM * 4);
  float*    z      = (float*)alloc((size_t)N_NODES * HDIM * 4);
  _Float16* z1f16  = (_Float16*)alloc((size_t)N_NODES * HDIM * 2);
  _Float16* wp0_1  = (_Float16*)alloc((size_t)HDIM * HDIM * 2);
  _Float16* wp0_2  = (_Float16*)alloc((size_t)HDIM * HDIM * 2);
  _Float16* wp1_1  = (_Float16*)alloc((size_t)HDIM * HDIM * 2);
  _Float16* wp1_2  = (_Float16*)alloc((size_t)HDIM * HDIM * 2);
  _Float16* wpg    = (_Float16*)alloc((size_t)HDIM * PHDIM * 2);
  float*    gate   = (float*)alloc((size_t)N_NODES * 4);
  float*    aexp   = (float*)alloc((size_t)N_NODES * 4);
  float*    mmax   = (float*)alloc((size_t)G_GRAPHS * 4);
  float*    ssum   = (float*)alloc((size_t)G_GRAPHS * 4);
  float*    pooled = (float*)alloc((size_t)G_GRAPHS * HDIM * 4);
  (void)ws_size; (void)in_sizes; (void)n_in; (void)out_size;

  // 1. pre-pack weights to WMMA B-fragment f16 layout
  pack_weight_kernel<<<256, 256, 0, stream>>>(l0_w1, wp0_1, HDIM, HDIM);
  pack_weight_kernel<<<256, 256, 0, stream>>>(l0_w2, wp0_2, HDIM, HDIM);
  pack_weight_kernel<<<256, 256, 0, stream>>>(l1_w1, wp1_1, HDIM, HDIM);
  pack_weight_kernel<<<256, 256, 0, stream>>>(l1_w2, wp1_2, HDIM, HDIM);
  pack_weight_kernel<<<128, 256, 0, stream>>>(gw1, wpg, HDIM, PHDIM);

  // 2. embed + concat
  embed_concat_kernel<<<(N_NODES * HDIM) / 256, 256, 0, stream>>>(x, emb, h);

  const int n4 = N_NODES * HDIM / 4;
  const int gemm_blocks = N_NODES / 32;  // 3125

  // 3. GIN block 0
  copy4_kernel<<<(n4 + 255) / 256, 256, 0, stream>>>((const float4*)h, (float4*)z, n4);
  edge_scatter_kernel<<<N_EDGES / 4, 256, 0, stream>>>(src, dst, h, z);
  gemm_relu_f16_kernel<<<gemm_blocks, 256, 0, stream>>>(z, wp0_1, l0_b1, z1f16);
  gemm_ln_res_kernel<<<gemm_blocks, 256, 0, stream>>>(z1f16, wp0_2, l0_b2, ln0_g, ln0_b, h);

  // 4. GIN block 1
  copy4_kernel<<<(n4 + 255) / 256, 256, 0, stream>>>((const float4*)h, (float4*)z, n4);
  edge_scatter_kernel<<<N_EDGES / 4, 256, 0, stream>>>(src, dst, h, z);
  gemm_relu_f16_kernel<<<gemm_blocks, 256, 0, stream>>>(z, wp1_1, l1_b1, z1f16);
  gemm_ln_res_kernel<<<gemm_blocks, 256, 0, stream>>>(z1f16, wp1_2, l1_b2, ln1_g, ln1_b, h);

  // 5. gate scores
  gate_kernel<<<gemm_blocks, 256, 0, stream>>>(h, wpg, gb1, gw2, gb2, gate);

  // 6. softmax-attention pooling
  pool_init_kernel<<<(G_GRAPHS * HDIM + 255) / 256, 256, 0, stream>>>(pooled, mmax, ssum);
  seg_max_kernel<<<(N_NODES + 255) / 256, 256, 0, stream>>>(gate, batch, mmax);
  seg_exp_kernel<<<(N_NODES + 255) / 256, 256, 0, stream>>>(gate, batch, mmax, aexp, ssum);
  pool_scatter_kernel<<<(N_NODES * (HDIM / 4) + 255) / 256, 256, 0, stream>>>(
      h, aexp, ssum, batch, pooled);

  // 7. classifier head
  cls_kernel<<<G_GRAPHS, PHDIM, 0, stream>>>(pooled, gfeat, cw1, cb1, cw2, cb2, out);
}
